// MultiHeadAttention_73083163509329
// MI455X (gfx1250) — compile-verified, matrix operand order checked by reading, not run on hardware
//
#include <hip/hip_runtime.h>

// MHA forward for MI455X (gfx1250, wave32, WMMA, TDM).
// Pipeline: 3x f16-WMMA projection GEMMs -> fused attention (scores in LDS,
// softmax, single attn write, V double-buffered into LDS by the Tensor Data
// Mover, P@V WMMA from LDS) -> f16-WMMA out-proj + residual -> LayerNorm.

typedef __attribute__((ext_vector_type(16))) _Float16 v16h;
typedef __attribute__((ext_vector_type(8)))  float    v8f;
typedef __attribute__((ext_vector_type(4)))  unsigned int u32x4;
typedef __attribute__((ext_vector_type(8)))  int      i32x8;
typedef __attribute__((ext_vector_type(4)))  int      i32x4;

constexpr int kB  = 4;
constexpr int kS  = 2048;
constexpr int kD  = 1024;
constexpr int kH  = 16;
constexpr int kDK = 64;
constexpr int kM  = kB * kS;          // 8192 token rows
constexpr float kScale = 0.125f;      // 1/sqrt(d_k)

// Attention LDS layout (dynamic shared, group segment starts at LDS addr 0):
constexpr unsigned kScOff   = 0;                        // 16 x kS f32 scores
constexpr unsigned kPOff    = 16 * kS * 4;              // 16 x kS f16 probs
constexpr unsigned kRedOff  = kPOff + 16 * kS * 2;      // 16 x 8 f32 scratch
constexpr unsigned kVbufOff = kRedOff + 512;            // V panels (TDM dest)
constexpr int      kVChunk  = 256;                      // keys per TDM panel
constexpr unsigned kVPanelB = kVChunk * kDK * 2;        // 32 KB per panel
constexpr unsigned kSmemSz  = kVbufOff + 2 * kVPanelB;  // ~257 KB (<320 KB WGP)

// ---- WMMA operand index helpers (ISA 7.12.2, wave32) -----------------------
// A (16x32 f16): lanes 0-15 -> M=lane, K in {0..7,16..23}; lanes 16-31 -> +8.
// B (32x16 f16): mirrored; C/D (16x16 f32): VGPR i -> M=i (+8 for hi lanes),
// N = lane&15.
__device__ __forceinline__ int lane_lo(int lane) { return lane & 15; }
__device__ __forceinline__ int lane_hi(int lane) { return (lane & 16) ? 8 : 0; }

#if __has_builtin(__builtin_amdgcn_tensor_load_to_lds)
// Issue one TDM tensor load: V panel [kVChunk keys x kDK dims] f16 -> LDS.
// D# per ISA 8.3/8.4 (2-D tensor, data_size=2B, dim0_stride = kDK).
__device__ __forceinline__ void tdm_load_v_panel(const _Float16* gsrc,
                                                 unsigned ldsOff)
{
    const unsigned long long gaddr = (unsigned long long)gsrc;
    u32x4 g0;
    g0.x = 1u;                                         // count=1 (user D#)
    g0.y = ldsOff;                                     // lds_addr
    g0.z = (unsigned)(gaddr & 0xffffffffu);            // global_addr[31:0]
    g0.w = (unsigned)((gaddr >> 32) & 0x01ffffffu)     // global_addr[56:32]
         | (2u << 30);                                 // type=2 ("image")
    i32x8 g1;
    g1[0] = 1 << 16;                                   // data_size=2B
    g1[1] = kDK << 16;                                 // tensor_dim0 = 64
    g1[2] = (kS & 0xffff) << 16;                       // tensor_dim1 = 2048
    g1[3] = kDK << 16;                                 // tile_dim0 = 64
    g1[4] = kVChunk;                                   // tile_dim1 = 256
    g1[5] = kDK;                                       // tensor_dim0_stride
    g1[6] = 0;
    g1[7] = 0;
    i32x4 gz = {0, 0, 0, 0};
#if __clang_major__ >= 23
    i32x8 gz8 = {0, 0, 0, 0, 0, 0, 0, 0};
    __builtin_amdgcn_tensor_load_to_lds(g0, g1, gz, gz, gz8, 0);
#else
    __builtin_amdgcn_tensor_load_to_lds(g0, g1, gz, gz, 0);
#endif
}
#endif

// ---------------------------------------------------------------------------
// Projection GEMM: Yh = headview(X @ W^T + bias), f16 out in [B,H,S,d_k].
// X (33 MB) + W (4 MB) fit in the 192 MB global L2, so operands stream from
// L2 without LDS staging; fp32 -> f16 conversion fused into the loads.
// ---------------------------------------------------------------------------
__global__ void __launch_bounds__(128)
qkv_proj_wmma(const float* __restrict__ X, const float* __restrict__ W,
              const float* __restrict__ bias, _Float16* __restrict__ Yh)
{
    const int lane = threadIdx.x & 31;
    const int wave = threadIdx.x >> 5;
    const int tile = blockIdx.x * 4 + wave;
    const int tn = tile & (kD / 16 - 1);
    const int tm = tile >> 6;
    const int m0 = tm * 16, n0 = tn * 16;
    const int klo = lane_hi(lane);
    const float* ap = X + (size_t)(m0 + lane_lo(lane)) * kD;
    const float* bp = W + (size_t)(n0 + lane_lo(lane)) * kD;

    v8f acc = {};
#pragma unroll 4
    for (int k0 = 0; k0 < kD; k0 += 32) {
        const float* a0 = ap + k0 + klo;
        const float* b0 = bp + k0 + klo;
        __builtin_prefetch(a0 + 128, 0, 0);     // speculative global_prefetch_b8
        __builtin_prefetch(b0 + 128, 0, 0);
        v16h a, b;
#pragma unroll
        for (int j = 0; j < 8; ++j) {
            a[j] = (_Float16)a0[j];  a[j + 8] = (_Float16)a0[j + 16];
            b[j] = (_Float16)b0[j];  b[j + 8] = (_Float16)b0[j + 16];
        }
        acc = __builtin_amdgcn_wmma_f32_16x16x32_f16(
            false, a, false, b, (short)0, acc, false, false);
    }

    const int n = n0 + lane_lo(lane);
    const float bn = bias[n];
    const int h = n >> 6, d = n & 63;
#pragma unroll
    for (int i = 0; i < 8; ++i) {
        const int m  = m0 + i + lane_hi(lane);
        const int b_ = m >> 11;
        const int s  = m & (kS - 1);
        const size_t off = (((size_t)(b_ * kH + h)) * kS + s) * kDK + d;
        Yh[off] = (_Float16)(acc[i] + bn);
    }
}

// ---------------------------------------------------------------------------
// Fused attention per (b,h,16-query block). Block = 128 threads = 4 waves.
// ---------------------------------------------------------------------------
__global__ void __launch_bounds__(128)
attn_softmax_wmma(const _Float16* __restrict__ Qh, const _Float16* __restrict__ Kh,
                  const _Float16* __restrict__ Vh, float* __restrict__ AttnOut,
                  float* __restrict__ Ctx)
{
    extern __shared__ char smem_raw[];
    float*    sc  = (float*)(smem_raw + kScOff);
    _Float16* P   = (_Float16*)(smem_raw + kPOff);
    float*    red = (float*)(smem_raw + kRedOff);

    const int tid  = threadIdx.x;
    const int lane = tid & 31;
    const int wave = tid >> 5;
    const int qblk = blockIdx.x & (kS / 16 - 1);
    const int bh   = blockIdx.x >> 7;
    const size_t base = (size_t)bh * kS * kDK;
    const int klo = lane_hi(lane);
    const int ll  = lane_lo(lane);

    // Q operand held in registers: 16x64 -> two 16x32 A fragments.
    const _Float16* qrow = Qh + base + (size_t)(qblk * 16 + ll) * kDK;
    v16h aq0, aq1;
#pragma unroll
    for (int j = 0; j < 8; ++j) {
        aq0[j]     = qrow[klo + j];        aq0[j + 8] = qrow[klo + 16 + j];
        aq1[j]     = qrow[32 + klo + j];   aq1[j + 8] = qrow[32 + klo + 16 + j];
    }

    // Phase 1: scores = (Q K^T) * scale, key tiles striped over waves.
    for (int kt = wave; kt < kS / 16; kt += 4) {
        const _Float16* krow = Kh + base + (size_t)(kt * 16 + ll) * kDK;
        v16h b0, b1;
#pragma unroll
        for (int j = 0; j < 8; ++j) {
            b0[j]     = krow[klo + j];        b0[j + 8] = krow[klo + 16 + j];
            b1[j]     = krow[32 + klo + j];   b1[j + 8] = krow[32 + klo + 16 + j];
        }
        v8f acc = {};
        acc = __builtin_amdgcn_wmma_f32_16x16x32_f16(
            false, aq0, false, b0, (short)0, acc, false, false);
        acc = __builtin_amdgcn_wmma_f32_16x16x32_f16(
            false, aq1, false, b1, (short)0, acc, false, false);
#pragma unroll
        for (int i = 0; i < 8; ++i) {
            const int r = i + lane_hi(lane);
            sc[r * kS + kt * 16 + ll] = acc[i] * kScale;
        }
    }
    __syncthreads();

    // Phase 2: row softmax (8 threads per row), write probs once, pack f16 P.
    const int r = tid >> 3, sub = tid & 7;
    float mx = -3.4e38f;
    for (int c = sub; c < kS; c += 8) mx = fmaxf(mx, sc[r * kS + c]);
    red[r * 8 + sub] = mx;
    __syncthreads();
    float rowmax = red[r * 8];
#pragma unroll
    for (int j = 1; j < 8; ++j) rowmax = fmaxf(rowmax, red[r * 8 + j]);
    __syncthreads();

    float sum = 0.f;
    for (int c = sub; c < kS; c += 8) {
        float e = __expf(sc[r * kS + c] - rowmax);
        sc[r * kS + c] = e;
        sum += e;
    }
    red[r * 8 + sub] = sum;
    __syncthreads();
    float tot = red[r * 8];
#pragma unroll
    for (int j = 1; j < 8; ++j) tot += red[r * 8 + j];
    const float inv = 1.f / tot;

    float* arow = AttnOut + ((size_t)bh * kS + qblk * 16 + r) * kS;
    for (int c = sub; c < kS; c += 8) {
        float p = sc[r * kS + c] * inv;
        arow[c] = p;                       // single mandatory 1.07 GB write
        P[r * kS + c] = (_Float16)p;
    }
    __syncthreads();

    // Phase 3: ctx[16x64] = P[16x2048] @ V[2048x64].
    // V double-buffered into LDS by the Tensor Data Mover (TENSORcnt
    // software pipeline); each wave owns a 16-wide d_k slice.
    const int n0 = wave * 16;
    v8f cacc = {};
    constexpr int nChunks = kS / kVChunk;

#if __has_builtin(__builtin_amdgcn_tensor_load_to_lds)
    if (wave == 0) {
        tdm_load_v_panel(Vh + base, kVbufOff);             // prime panel 0
    }
#endif
    for (int chunk = 0; chunk < nChunks; ++chunk) {
        const _Float16* Vb =
            (_Float16*)(smem_raw + kVbufOff + (chunk & 1) * kVPanelB);
#if __has_builtin(__builtin_amdgcn_tensor_load_to_lds)
        if (wave == 0) {
            if (chunk + 1 < nChunks) {
                tdm_load_v_panel(
                    Vh + base + (size_t)(chunk + 1) * kVChunk * kDK,
                    kVbufOff + ((chunk + 1) & 1) * kVPanelB);
                __builtin_amdgcn_s_wait_tensorcnt(1);  // panel `chunk` resident
            } else {
                __builtin_amdgcn_s_wait_tensorcnt(0);
            }
        }
#else
        // Fallback: cooperative vector copy of the V panel.
        {
            const _Float16* src = Vh + base + (size_t)chunk * kVChunk * kDK;
            _Float16* dst = (_Float16*)(smem_raw + kVbufOff
                                        + (chunk & 1) * kVPanelB);
            for (int t = tid * 8; t < kVChunk * kDK; t += 128 * 8) {
                *(uint4*)&dst[t] = *(const uint4*)&src[t];
            }
        }
#endif
        __syncthreads();

#pragma unroll
        for (int kk = 0; kk < kVChunk; kk += 32) {
            v16h a, b;
#pragma unroll
            for (int j = 0; j < 8; ++j) {
                a[j]     = P[ll * kS + chunk * kVChunk + kk + klo + j];
                a[j + 8] = P[ll * kS + chunk * kVChunk + kk + klo + 16 + j];
                b[j]     = Vb[(kk + klo + j)      * kDK + n0 + ll];
                b[j + 8] = Vb[(kk + klo + 16 + j) * kDK + n0 + ll];
            }
            cacc = __builtin_amdgcn_wmma_f32_16x16x32_f16(
                false, a, false, b, (short)0, cacc, false, false);
        }
        __syncthreads();   // panel consumed before its buffer is re-targeted
    }

    const int b_ = bh >> 4, h = bh & 15;
#pragma unroll
    for (int i = 0; i < 8; ++i) {
        const int srow = qblk * 16 + i + lane_hi(lane);
        Ctx[((size_t)b_ * kS + srow) * kD + h * kDK + n0 + ll] = cacc[i];
    }
}

// ---------------------------------------------------------------------------
// Output projection + residual: X = Ctx @ Wo^T + bo + Resid (fp32 out).
// ---------------------------------------------------------------------------
__global__ void __launch_bounds__(128)
out_proj_wmma(const float* __restrict__ Ctx, const float* __restrict__ Wo,
              const float* __restrict__ bo, const float* __restrict__ Resid,
              float* __restrict__ Xout)
{
    const int lane = threadIdx.x & 31;
    const int wave = threadIdx.x >> 5;
    const int tile = blockIdx.x * 4 + wave;
    const int tn = tile & (kD / 16 - 1);
    const int tm = tile >> 6;
    const int m0 = tm * 16, n0 = tn * 16;
    const int klo = lane_hi(lane);
    const float* ap = Ctx + (size_t)(m0 + lane_lo(lane)) * kD;
    const float* bp = Wo  + (size_t)(n0 + lane_lo(lane)) * kD;

    v8f acc = {};
#pragma unroll 4
    for (int k0 = 0; k0 < kD; k0 += 32) {
        const float* a0 = ap + k0 + klo;
        const float* b0 = bp + k0 + klo;
        __builtin_prefetch(a0 + 128, 0, 0);
        __builtin_prefetch(b0 + 128, 0, 0);
        v16h a, b;
#pragma unroll
        for (int j = 0; j < 8; ++j) {
            a[j] = (_Float16)a0[j];  a[j + 8] = (_Float16)a0[j + 16];
            b[j] = (_Float16)b0[j];  b[j + 8] = (_Float16)b0[j + 16];
        }
        acc = __builtin_amdgcn_wmma_f32_16x16x32_f16(
            false, a, false, b, (short)0, acc, false, false);
    }

    const int n = n0 + lane_lo(lane);
    const float bn = bo[n];
#pragma unroll
    for (int i = 0; i < 8; ++i) {
        const int m = m0 + i + lane_hi(lane);
        const size_t off = (size_t)m * kD + n;
        Xout[off] = acc[i] + bn + Resid[off];
    }
}

// ---------------------------------------------------------------------------
// Row LayerNorm over D=1024 (one block per token row).
// ---------------------------------------------------------------------------
__global__ void __launch_bounds__(256)
layernorm_kernel(const float* __restrict__ X, const float* __restrict__ gamma,
                 const float* __restrict__ beta, float* __restrict__ Out)
{
    __shared__ float rbuf[256];
    const int row = blockIdx.x, tid = threadIdx.x;
    const float* xr = X + (size_t)row * kD;

    float s = 0.f;
    for (int i = tid; i < kD; i += 256) s += xr[i];
    rbuf[tid] = s; __syncthreads();
    for (int off = 128; off > 0; off >>= 1) {
        if (tid < off) rbuf[tid] += rbuf[tid + off];
        __syncthreads();
    }
    const float mu = rbuf[0] * (1.f / kD);
    __syncthreads();

    float vs = 0.f;
    for (int i = tid; i < kD; i += 256) { float dd = xr[i] - mu; vs += dd * dd; }
    rbuf[tid] = vs; __syncthreads();
    for (int off = 128; off > 0; off >>= 1) {
        if (tid < off) rbuf[tid] += rbuf[tid + off];
        __syncthreads();
    }
    const float inv = rsqrtf(rbuf[0] * (1.f / kD) + 1e-5f);

    for (int i = tid; i < kD; i += 256)
        Out[(size_t)row * kD + i] = (xr[i] - mu) * inv * gamma[i] + beta[i];
}

// ---------------------------------------------------------------------------
extern "C" void kernel_launch(void* const* d_in, const int* in_sizes, int n_in,
                              void* d_out, int out_size, void* d_ws, size_t ws_size,
                              hipStream_t stream)
{
    (void)in_sizes; (void)n_in; (void)out_size; (void)ws_size;

    const float* q   = (const float*)d_in[0];
    const float* k   = (const float*)d_in[1];
    const float* v   = (const float*)d_in[2];
    const float* w_q = (const float*)d_in[3];
    const float* b_q = (const float*)d_in[4];
    const float* w_k = (const float*)d_in[5];
    const float* b_k = (const float*)d_in[6];
    const float* w_v = (const float*)d_in[7];
    const float* b_v = (const float*)d_in[8];
    const float* w_o = (const float*)d_in[9];
    const float* b_o = (const float*)d_in[10];
    const float* lng = (const float*)d_in[11];
    const float* lnb = (const float*)d_in[12];

    float* out  = (float*)d_out;                   // [B,S,D]
    float* attn = out + (size_t)kM * kD;           // [B,H,S,S] follows

    _Float16* qh = (_Float16*)d_ws;
    _Float16* kh = qh + (size_t)kM * kD;
    _Float16* vh = kh + (size_t)kM * kD;
    float* ctx  = (float*)(vh + (size_t)kM * kD);
    float* xbuf = ctx + (size_t)kM * kD;

    const dim3 blk(128);
    const dim3 gproj((kM / 16) * (kD / 16) / 4);

    qkv_proj_wmma<<<gproj, blk, 0, stream>>>(q, w_q, b_q, qh);
    qkv_proj_wmma<<<gproj, blk, 0, stream>>>(k, w_k, b_k, kh);
    qkv_proj_wmma<<<gproj, blk, 0, stream>>>(v, w_v, b_v, vh);

    attn_softmax_wmma<<<dim3(kB * kH * (kS / 16)), blk, kSmemSz, stream>>>(
        qh, kh, vh, attn, ctx);

    out_proj_wmma<<<gproj, blk, 0, stream>>>(ctx, w_o, b_o, q, xbuf);

    layernorm_kernel<<<dim3(kM), dim3(256), 0, stream>>>(xbuf, lng, lnb, out);
}
